// MultiHeadedAttention_50826642981217
// MI455X (gfx1250) — matrix operand order repaired and checked by reading, not compiled, not run on hardware
//
#include <hip/hip_runtime.h>

typedef _Float16 v16h __attribute__((ext_vector_type(16)));
typedef _Float16 v8h  __attribute__((ext_vector_type(8)));
typedef float    v8f  __attribute__((ext_vector_type(8)));
typedef int      i32x4 __attribute__((vector_size(16)));
typedef __attribute__((address_space(1))) i32x4 as1_i32x4;   // global (prints as __device__)
typedef __attribute__((address_space(3))) i32x4 as3_i32x4;   // LDS    (prints as __shared__)

#define BB   2
#define SQ   2048
#define HIDD 1024
#define NHH  16
#define DKK  64

// ---- gfx1250 async global->LDS path (guarded; falls back to sync staging) --
#if defined(__has_builtin)
#if __has_builtin(__builtin_amdgcn_global_load_async_to_lds_b128) && \
    __has_builtin(__builtin_amdgcn_s_wait_asynccnt)
#define HAVE_ASYNC 1
#endif
#endif
#ifndef HAVE_ASYNC
#define HAVE_ASYNC 0
#endif

#if HAVE_ASYNC
#define WAIT_ASYNC(n) __builtin_amdgcn_s_wait_asynccnt(n)
#else
#define WAIT_ASYNC(n)
#endif

__device__ __forceinline__ void copy16(const _Float16* g, _Float16* l) {
#if HAVE_ASYNC
    __builtin_amdgcn_global_load_async_to_lds_b128(
        (as1_i32x4*)(const_cast<_Float16*>(g)),
        (as3_i32x4*)(l), 0, 0);
#else
    *(v8h*)l = *(const v8h*)g;
#endif
}

union V16U { v16h v; v8h h[2]; };

__device__ __forceinline__ v16h cat8(v8h lo, v8h hi) {
    V16U u; u.h[0] = lo; u.h[1] = hi; return u.v;
}

__device__ __forceinline__ v8f wmma_f16(v16h a, v16h b, v8f c) {
    return __builtin_amdgcn_wmma_f32_16x16x32_f16(
        false, a, false, b, (short)0, c, false, false);
}

// ---------------------------------------------------------------------------
// Convert inputs to f16; fold key/value scaling by (1 + bias[b,s]).
// ---------------------------------------------------------------------------
__global__ __launch_bounds__(256)
void cvt_inputs(const float* __restrict__ q, const float* __restrict__ k,
                const float* __restrict__ v, const float* __restrict__ bias,
                _Float16* __restrict__ qh, _Float16* __restrict__ kh,
                _Float16* __restrict__ vh) {
    int i = blockIdx.x * 256 + threadIdx.x;
    int row = i >> 10;
    float sc = 1.0f + bias[row];
    qh[i] = (_Float16)q[i];
    kh[i] = (_Float16)(k[i] * sc);
    vh[i] = (_Float16)(v[i] * sc);
}

__global__ __launch_bounds__(256)
void cvt_w(const float* __restrict__ w, _Float16* __restrict__ wh) {
    int i = blockIdx.x * 256 + threadIdx.x;
    wh[i] = (_Float16)w[i];
}

// ---------------------------------------------------------------------------
// NT GEMM via WMMA:  Y[m,n] = sum_k A[m,k] * W[n,k] + bias[n]
// MODE 0: f16 (B,NH,S,DK) | MODE 1: f16 (B,NH,DK,S) | MODE 2: f32 (M,N)
// ---------------------------------------------------------------------------
template <int MODE>
__global__ __launch_bounds__(256)
void gemm_wmma(const _Float16* __restrict__ A, const _Float16* __restrict__ W,
               const float* __restrict__ bias, void* __restrict__ out) {
    const int K = 1024, N = 1024;
    int lane = threadIdx.x & 31, wave = threadIdx.x >> 5;
    int n15 = lane & 15, h = lane >> 4;
    int m0 = blockIdx.x * 64 + (wave & 3) * 16;
    int n0 = blockIdx.y * 128 + (wave >> 2) * 64;

    v8f acc[4] = { {}, {}, {}, {} };
    const _Float16* arow = A + (size_t)(m0 + n15) * K;

    for (int k0 = 0; k0 < K; k0 += 32) {
        v8h alo = *(const v8h*)(arow + k0 + 8 * h);
        v8h ahi = *(const v8h*)(arow + k0 + 16 + 8 * h);
        v16h a = cat8(alo, ahi);
#pragma unroll
        for (int nq = 0; nq < 4; ++nq) {
            const _Float16* wrow = W + (size_t)(n0 + nq * 16 + n15) * K + k0;
            v16h b = cat8(*(const v8h*)(wrow + 8 * h), *(const v8h*)(wrow + 16 + 8 * h));
            acc[nq] = wmma_f16(a, b, acc[nq]);
        }
    }

#pragma unroll
    for (int nq = 0; nq < 4; ++nq) {
        int col = n0 + nq * 16 + n15;
        float bc = bias[col];
#pragma unroll
        for (int r = 0; r < 8; ++r) {
            float y = acc[nq][r] + bc;
            int m = m0 + r + 8 * h;
            if (MODE == 2) {
                ((float*)out)[(size_t)m * N + col] = y;
            } else {
                int b_ = m >> 11, s_ = m & (SQ - 1);
                int hh = col >> 6, d = col & (DKK - 1);
                if (MODE == 0)
                    ((_Float16*)out)[(((size_t)(b_ * NHH + hh)) * SQ + s_) * DKK + d] = (_Float16)y;
                else
                    ((_Float16*)out)[(((size_t)(b_ * NHH + hh)) * DKK + d) * SQ + s_] = (_Float16)y;
            }
        }
    }
}

// ---------------------------------------------------------------------------
// Flash attention: grid (B*NH, S/128), 8 waves/WG, 16 query rows per wave.
// K/V 32-key tiles staged in LDS, double-buffered via async-to-LDS (ASYNCcnt).
// Row-sum of P computed by WMMA against an all-ones B matrix (no shuffles).
// ---------------------------------------------------------------------------
__global__ __launch_bounds__(256)
void flash_attn(const _Float16* __restrict__ Qp,   // (B,NH,S,DK)
                const _Float16* __restrict__ Kp,   // (B,NH,S,DK)
                const _Float16* __restrict__ Vpt,  // (B,NH,DK,S)
                const int* __restrict__ mask,      // (B,1,S)
                _Float16* __restrict__ X) {        // (B,S,HID) f16
    __shared__ _Float16 Kst[2 * 32 * 72];   // K tile: 32 rows, stride 72 halves
    __shared__ _Float16 Vst[2 * 64 * 40];   // V^T tile: 64 rows, stride 40 halves
    __shared__ _Float16 Pst[8 * 16 * 40];   // per-wave P tile, stride 40

    int t = threadIdx.x;
    int lane = t & 31, wave = t >> 5;
    int n15 = lane & 15, h = lane >> 4;
    int bh = blockIdx.x;
    int b = bh >> 4, hh = bh & 15;
    int q0 = blockIdx.y * 128 + wave * 16;

    const _Float16* Kbh = Kp + (size_t)bh * SQ * DKK;
    const _Float16* Vbh = Vpt + (size_t)bh * DKK * SQ;
    const int* mrow = mask + b * SQ;

    // staging coords: 256 threads x 16B cover each 4KB tile
    int krow = t >> 3, kc = (t & 7) * 8;    // K: 32 rows x 128B
    int vrow = t >> 2, vc = (t & 3) * 8;    // V: 64 rows x 64B

    // Q A-fragments (16 x 64 = two K=32 chunks)
    const _Float16* Qrow = Qp + ((size_t)bh * SQ + q0 + n15) * DKK;
    v16h qa[2];
#pragma unroll
    for (int c = 0; c < 2; ++c)
        qa[c] = cat8(*(const v8h*)(Qrow + c * 32 + 8 * h),
                     *(const v8h*)(Qrow + c * 32 + 16 + 8 * h));

    _Float16* pw = Pst + wave * 640;
    const v16h ones = { (_Float16)1, (_Float16)1, (_Float16)1, (_Float16)1,
                        (_Float16)1, (_Float16)1, (_Float16)1, (_Float16)1,
                        (_Float16)1, (_Float16)1, (_Float16)1, (_Float16)1,
                        (_Float16)1, (_Float16)1, (_Float16)1, (_Float16)1 };

    v8f o[4] = { {}, {}, {}, {} };
    float mreg[8], lreg[8], alpha[8];
#pragma unroll
    for (int r = 0; r < 8; ++r) { mreg[r] = -1e30f; lreg[r] = 0.0f; }
    const float sc2   = 0.125f * 1.44269504088896340736f;  // (1/sqrt(64))*log2(e)
    const float MASKV = -10000.0f * 1.44269504088896340736f;

    // prologue: stage tile 0 into buffer 0
    copy16(Kbh + (size_t)krow * 64 + kc, Kst + krow * 72 + kc);
    copy16(Vbh + (size_t)vrow * SQ + vc, Vst + vrow * 40 + vc);

    const int NT = SQ / 32;
    for (int it = 0; it < NT; ++it) {
        int kk = it * 32;
        int cur = it & 1;
        __syncthreads();                    // everyone done reading buffer cur^1
        if (it + 1 < NT) {
            int kn = kk + 32;
            _Float16* Kd = Kst + (cur ^ 1) * (32 * 72);
            _Float16* Vd = Vst + (cur ^ 1) * (64 * 40);
            copy16(Kbh + (size_t)(kn + krow) * 64 + kc, Kd + krow * 72 + kc);
            copy16(Vbh + (size_t)vrow * SQ + kn + vc, Vd + vrow * 40 + vc);
            WAIT_ASYNC(2);                  // tile it complete (in-order), it+1 in flight
        } else {
            WAIT_ASYNC(0);
        }
        __syncthreads();                    // tile it visible to all waves
        const _Float16* Kl = Kst + cur * (32 * 72);
        const _Float16* Vl = Vst + cur * (64 * 40);

        // ---- scores: S = Q * K^T, two 16-key fragments -------------------
        v8f s0 = {}, s1 = {};
#pragma unroll
        for (int c = 0; c < 2; ++c) {
            v16h b0 = cat8(*(const v8h*)(Kl + n15 * 72 + c * 32 + 8 * h),
                           *(const v8h*)(Kl + n15 * 72 + c * 32 + 16 + 8 * h));
            v16h b1 = cat8(*(const v8h*)(Kl + (16 + n15) * 72 + c * 32 + 8 * h),
                           *(const v8h*)(Kl + (16 + n15) * 72 + c * 32 + 16 + 8 * h));
            s0 = wmma_f16(qa[c], b0, s0);
            s1 = wmma_f16(qa[c], b1, s1);
        }
        bool mk0 = mrow[kk + n15] != 0;
        bool mk1 = mrow[kk + 16 + n15] != 0;

        // ---- online softmax in log2 domain (row = r + 8h) ----------------
#pragma unroll
        for (int r = 0; r < 8; ++r) {
            float v0 = mk0 ? s0[r] * sc2 : MASKV;
            float v1 = mk1 ? s1[r] * sc2 : MASKV;
            float tm = fmaxf(v0, v1);
#pragma unroll
            for (int off = 1; off < 16; off <<= 1) tm = fmaxf(tm, __shfl_xor(tm, off, 32));
            float mn = fmaxf(mreg[r], tm);
            alpha[r] = exp2f(mreg[r] - mn);
            mreg[r] = mn;
            pw[(r + 8 * h) * 40 + n15]      = (_Float16)exp2f(v0 - mn);
            pw[(r + 8 * h) * 40 + 16 + n15] = (_Float16)exp2f(v1 - mn);
        }
        __builtin_amdgcn_wave_barrier();    // DS ops are wave-in-order

        // ---- reload P in A-fragment layout -------------------------------
        v16h pa = cat8(*(const v8h*)(pw + n15 * 40 + 8 * h),
                       *(const v8h*)(pw + n15 * 40 + 16 + 8 * h));
        __builtin_amdgcn_wave_barrier();

        // ---- row-sum of P via WMMA with ones matrix ----------------------
        v8f z = {};
        v8f lsum = wmma_f16(pa, ones, z);

        // ---- O = O*alpha + P*V -------------------------------------------
#pragma unroll
        for (int nq = 0; nq < 4; ++nq) {
            v8f oo = o[nq];
#pragma unroll
            for (int r = 0; r < 8; ++r) oo[r] *= alpha[r];
            v16h vb = cat8(*(const v8h*)(Vl + (nq * 16 + n15) * 40 + 8 * h),
                           *(const v8h*)(Vl + (nq * 16 + n15) * 40 + 16 + 8 * h));
            o[nq] = wmma_f16(pa, vb, oo);
        }
#pragma unroll
        for (int r = 0; r < 8; ++r) lreg[r] = lreg[r] * alpha[r] + lsum[r];
    }

    // ---- finalize: divide by l, store (B,S,HID) f16 ----------------------
    _Float16* Xb = X + ((size_t)b * SQ + q0) * HIDD + hh * DKK;
#pragma unroll
    for (int nq = 0; nq < 4; ++nq)
#pragma unroll
        for (int r = 0; r < 8; ++r)
            Xb[(size_t)(r + 8 * h) * HIDD + nq * 16 + n15] = (_Float16)(o[nq][r] / lreg[r]);
}

// ---------------------------------------------------------------------------
extern "C" void kernel_launch(void* const* d_in, const int* in_sizes, int n_in,
                              void* d_out, int out_size, void* d_ws, size_t ws_size,
                              hipStream_t stream) {
    (void)in_sizes; (void)n_in; (void)out_size; (void)ws_size;
    const float* query = (const float*)d_in[0];
    const float* key_  = (const float*)d_in[1];
    const float* value = (const float*)d_in[2];
    const float* bias  = (const float*)d_in[3];
    const int*   mask  = (const int*)d_in[4];
    const float* wq = (const float*)d_in[5];
    const float* bq = (const float*)d_in[6];
    const float* wk = (const float*)d_in[7];
    const float* bk = (const float*)d_in[8];
    const float* wv = (const float*)d_in[9];
    const float* bv = (const float*)d_in[10];
    const float* wo = (const float*)d_in[11];
    const float* bo = (const float*)d_in[12];

    char* ws = (char*)d_ws;
    const size_t SEG = (size_t)4096 * 1024 * sizeof(_Float16);  // 8 MB
    _Float16* qh  = (_Float16*)(ws + 0 * SEG);
    _Float16* kh  = (_Float16*)(ws + 1 * SEG);
    _Float16* vh  = (_Float16*)(ws + 2 * SEG);
    _Float16* Qp  = (_Float16*)(ws + 3 * SEG);
    _Float16* Kp  = (_Float16*)(ws + 4 * SEG);
    _Float16* Vpt = (_Float16*)(ws + 5 * SEG);
    _Float16* Wh  = (_Float16*)(ws + 6 * SEG);                  // 4 x 2 MB
    _Float16* wqh = Wh;
    _Float16* wkh = Wh + 1024 * 1024;
    _Float16* wvh = Wh + 2 * 1024 * 1024;
    _Float16* woh = Wh + 3 * 1024 * 1024;
    _Float16* Xh  = qh;   // reuse: q-projection consumed qh before flash writes

    cvt_inputs<<<16384, 256, 0, stream>>>(query, key_, value, bias, qh, kh, vh);
    cvt_w<<<4096, 256, 0, stream>>>(wq, wqh);
    cvt_w<<<4096, 256, 0, stream>>>(wk, wkh);
    cvt_w<<<4096, 256, 0, stream>>>(wv, wvh);
    cvt_w<<<4096, 256, 0, stream>>>(wo, woh);

    dim3 g(64, 8);   // M/64 x N/128
    gemm_wmma<0><<<g, 256, 0, stream>>>(qh, wqh, bq, Qp);
    gemm_wmma<0><<<g, 256, 0, stream>>>(kh, wkh, bk, Kp);
    gemm_wmma<1><<<g, 256, 0, stream>>>(vh, wvh, bv, Vpt);

    dim3 fa(BB * NHH, SQ / 128);
    flash_attn<<<fa, 256, 0, stream>>>(Qp, Kp, Vpt, mask, Xh);

    gemm_wmma<2><<<g, 256, 0, stream>>>(Xh, woh, bo, d_out);
}
